// Tramba_57183194579326
// MI455X (gfx1250) — compile-verified, matrix operand order checked by reading, not run on hardware
//
#include <hip/hip_runtime.h>

typedef __attribute__((ext_vector_type(16))) _Float16 v16h;
typedef __attribute__((ext_vector_type(8)))  float    v8f;

#define B_   2
#define T_   288
#define L_   366
#define D_   32
#define ED_  64
#define DS_  16
#define DC_  4
#define NSEQ (B_ * L_)      // 732
#define CH   16             // timesteps per chunk
#define NCH  (T_ / CH)      // 18
#define DBCW 34             // DT_RANK + 2*DS

__device__ __forceinline__ float fast_sigmoid(float x) {
    return __builtin_amdgcn_rcpf(1.f + __expf(-x));
}

// ---------------------------------------------------------------------------
// Stage 1: h = ((x @ emb_w + emb_b) + pos_emb[t]) @ agg_w + agg_b,
// written transposed into XM[(b*L+l)*T + t][d]. One wave per (b,t,l).
// ---------------------------------------------------------------------------
__global__ void __launch_bounds__(256) embed_agg_kernel(
    const float* __restrict__ x, const float* __restrict__ emb_w,
    const float* __restrict__ emb_b, const float* __restrict__ pos_emb,
    const float* __restrict__ agg_w, const float* __restrict__ agg_b,
    float* __restrict__ XM)
{
    __shared__ float s_aggw[D_ * D_];
    __shared__ float s_embw[3 * D_];
    __shared__ float s_embb[D_];
    __shared__ float s_aggb[D_];
    const int tid = threadIdx.x, lane = tid & 31, wave = tid >> 5;
    for (int i = tid; i < D_ * D_; i += 256) s_aggw[i] = agg_w[i];
    if (tid < 3 * D_) s_embw[tid] = emb_w[tid];
    if (tid < D_) { s_embb[tid] = emb_b[tid]; s_aggb[tid] = agg_b[tid]; }
    __syncthreads();

    const long pos = (long)blockIdx.x * 8 + wave;     // (b*T + t)*L + l
    if (pos >= (long)B_ * T_ * L_) return;
    const int l = (int)(pos % L_);
    const long bt = pos / L_;
    const int t = (int)(bt % T_);
    const int b = (int)(bt / T_);

    const float* xp = x + pos * 3;
    const float x0 = xp[0], x1 = xp[1], x2 = xp[2];
    const int d = lane;
    float e = x0 * s_embw[d] + x1 * s_embw[D_ + d] + x2 * s_embw[2 * D_ + d]
            + s_embb[d] + pos_emb[t * D_ + d];
    float h = s_aggb[d];
#pragma unroll
    for (int k = 0; k < D_; ++k)
        h += __shfl(e, k, 32) * s_aggw[k * D_ + d];

    XM[((long)(b * L_ + l) * T_ + t) * D_ + d] = h;
}

// ---------------------------------------------------------------------------
// Stage 2: one fused Mamba layer, one workgroup (256 threads) per sequence.
// In-place residual update of XM. WMMA (f32<-f16) for in/out projections.
// Weight matrices are pre-swizzled into per-lane WMMA B-fragment order so
// each fragment is two ds_load_b128's.
// ---------------------------------------------------------------------------
__global__ void __launch_bounds__(256) mamba_layer_kernel(
    float* __restrict__ XM,
    const float* __restrict__ norm_w, const float* __restrict__ in_w,
    const float* __restrict__ conv_w, const float* __restrict__ conv_b,
    const float* __restrict__ x_w,    const float* __restrict__ dt_w,
    const float* __restrict__ dt_b,   const float* __restrict__ A_log,
    const float* __restrict__ Dp,     const float* __restrict__ out_w)
{
    // B fragments in per-lane order: [(tile)*32 + lane]*16 + j, 32B aligned.
    __shared__ __attribute__((aligned(32))) _Float16 s_inwf[8 * 32 * 16];   // 8KB
    __shared__ __attribute__((aligned(32))) _Float16 s_outwf[4 * 32 * 16];  // 4KB
    __shared__ float    s_convw[ED_ * DC_];
    __shared__ float    s_convb[ED_];
    __shared__ float    s_xw[ED_ * DBCW];
    __shared__ float    s_dtw[2 * ED_];
    __shared__ float    s_dtb[ED_];
    __shared__ float    s_xch[CH * D_];           // residual chunk (f32)
    __shared__ __attribute__((aligned(32))) _Float16 s_xnc[CH * D_]; // rms chunk f16
    __shared__ float    s_xz[CH * 2 * ED_];       // in-proj output
    __shared__ float    s_hist[(DC_ - 1) * ED_];  // conv history rows t-3..t-1
    __shared__ float    s_xc[CH * ED_];           // conv+silu
    __shared__ float    s_dbc[CH * DBCW];
    __shared__ float    s_delta[CH * ED_];
    __shared__ float    s_y[CH * ED_];
    __shared__ __attribute__((aligned(32))) _Float16 s_yg[CH * ED_]; // gated y f16

    const int tid = threadIdx.x, lane = tid & 31, wave = tid >> 5;
    float* xm = XM + (long)blockIdx.x * T_ * D_;

    // ---- weight staging: swizzle in_w (32x128) into in-proj B fragments ----
    for (int i = tid; i < 8 * 32 * 16; i += 256) {
        const int j  = i & 15;
        const int ln = (i >> 4) & 31;
        const int w  = i >> 9;
        const int col = w * 16 + (ln & 15);
        const int kb2 = (ln < 16) ? 0 : 8;
        const int k = kb2 + j + ((j >= 8) ? 8 : 0);   // ISA 16-bit 32xN layout
        s_inwf[i] = (_Float16)in_w[k * (2 * ED_) + col];
    }
    // ---- swizzle out_w (64x32) into out-proj B fragments (kc,wave tiles) ----
    for (int i = tid; i < 4 * 32 * 16; i += 256) {
        const int j  = i & 15;
        const int ln = (i >> 4) & 31;
        const int wk = i >> 9;          // kc*2 + w
        const int w  = wk & 1, kc = wk >> 1;
        const int col = w * 16 + (ln & 15);
        const int kb2 = (ln < 16) ? 0 : 8;
        const int k = kc * 32 + kb2 + j + ((j >= 8) ? 8 : 0);
        s_outwf[i] = (_Float16)out_w[k * D_ + col];
    }
    for (int i = tid; i < ED_ * DC_;  i += 256) s_convw[i] = conv_w[i];
    for (int i = tid; i < ED_ * DBCW; i += 256) s_xw[i] = x_w[i];
    if (tid < 2 * ED_) s_dtw[tid] = dt_w[tid];
    if (tid < ED_) { s_convb[tid] = conv_b[tid]; s_dtb[tid] = dt_b[tid]; }
    if (tid < (DC_ - 1) * ED_) s_hist[tid] = 0.f;
    const float nwv = norm_w[lane];               // lane == d when used

    // per-channel scan constants + persistent state in registers (e = tid)
    float Ae[DS_], hst[DS_], dpv = 0.f;
#pragma unroll
    for (int s = 0; s < DS_; ++s) hst[s] = 0.f;
    if (tid < ED_) {
        dpv = Dp[tid];
#pragma unroll
        for (int s = 0; s < DS_; ++s) Ae[s] = -__expf(A_log[tid * DS_ + s]);
    }
    __syncthreads();

    for (int c = 0; c < NCH; ++c) {
        const int t0 = c * CH;
        if (c + 1 < NCH) __builtin_prefetch(xm + (t0 + CH) * D_, 0, 1);

        // ---- load residual chunk ----
        for (int i = tid; i < CH * D_; i += 256) s_xch[i] = xm[t0 * D_ + i];
        __syncthreads();

        // ---- RMS norm (wave per row) -> f16 ----
        for (int r = wave; r < CH; r += 8) {
            float v = s_xch[r * D_ + lane];
            float ss = v * v;
#pragma unroll
            for (int off = 16; off; off >>= 1) ss += __shfl_xor(ss, off, 32);
            float rinv = rsqrtf(ss * (1.0f / D_) + 1e-5f);
            s_xnc[r * D_ + lane] = (_Float16)(v * rinv * nwv);
        }
        __syncthreads();

        // ---- in-projection: (16x32) @ (32x128), wave w -> N-tile w ----
        {
            const int col  = wave * 16 + (lane & 15);
            const int mrow = lane & 15;
            const int kb   = (lane < 16) ? 0 : 8;   // ISA 16-bit A 16x32 layout
            v16h a;
#pragma unroll
            for (int j = 0; j < 8; ++j) {
                a[j]     = s_xnc[mrow * D_ + kb + j];
                a[8 + j] = s_xnc[mrow * D_ + kb + 16 + j];
            }
            const v16h bfrag = *(const v16h*)&s_inwf[(wave * 32 + lane) * 16];
            v8f acc = {};
            acc = __builtin_amdgcn_wmma_f32_16x16x32_f16(
                false, a, false, bfrag, (short)0, acc, false, false);
            const int rbase = (lane < 16) ? 0 : 8;
#pragma unroll
            for (int vr = 0; vr < 8; ++vr)
                s_xz[(vr + rbase) * (2 * ED_) + col] = acc[vr];
        }
        __syncthreads();

        // ---- depthwise conv (k=4, causal) + SiLU ----
        for (int idx = tid; idx < CH * ED_; idx += 256) {
            const int t = idx >> 6, e = idx & 63;
            float acc = s_convb[e];
#pragma unroll
            for (int k = 0; k < DC_; ++k) {
                const int tt = t - (DC_ - 1) + k;
                float xv;
                if (t0 + tt < 0)      xv = 0.f;
                else if (tt < 0)      xv = s_hist[(tt + DC_ - 1) * ED_ + e];
                else                  xv = s_xz[tt * (2 * ED_) + e];
                acc += xv * s_convw[e * DC_ + k];
            }
            s_xc[idx] = acc * fast_sigmoid(acc);
        }
        __syncthreads();

        // ---- conv history update + dbc = xin @ x_w (64 -> 34) ----
        for (int idx = tid; idx < (DC_ - 1) * ED_; idx += 256) {
            const int r = idx >> 6, e = idx & 63;
            s_hist[idx] = s_xz[(CH - (DC_ - 1) + r) * (2 * ED_) + e];
        }
        for (int idx = tid; idx < CH * DBCW; idx += 256) {
            const int t = idx / DBCW, j = idx % DBCW;
            float a = 0.f;
            for (int e = 0; e < ED_; ++e) a += s_xc[t * ED_ + e] * s_xw[e * DBCW + j];
            s_dbc[idx] = a;
        }
        __syncthreads();

        // ---- delta = softplus(dbc[:,:2] @ dt_w + dt_b) ----
        for (int idx = tid; idx < CH * ED_; idx += 256) {
            const int t = idx >> 6, e = idx & 63;
            float pre = s_dbc[t * DBCW] * s_dtw[e]
                      + s_dbc[t * DBCW + 1] * s_dtw[ED_ + e] + s_dtb[e];
            s_delta[idx] = (pre > 20.f) ? pre : __logf(1.f + __expf(pre));
        }
        __syncthreads();

        // ---- selective scan (threads 0..63, one ED channel each) ----
        if (tid < ED_) {
            const int e = tid;
            for (int t = 0; t < CH; ++t) {
                const float de = s_delta[t * ED_ + e];
                const float u  = s_xc[t * ED_ + e];
                const float du = de * u;
                const float* Bt = &s_dbc[t * DBCW + 2];
                const float* Ct = &s_dbc[t * DBCW + 2 + DS_];
                float yv = 0.f;
#pragma unroll
                for (int s = 0; s < DS_; ++s) {
                    float hv = __expf(de * Ae[s]) * hst[s] + du * Bt[s];
                    hst[s] = hv;
                    yv += hv * Ct[s];
                }
                s_y[t * ED_ + e] = yv + dpv * u;
            }
        }
        __syncthreads();

        // ---- gate: yg = y * silu(z) ----
        for (int idx = tid; idx < CH * ED_; idx += 256) {
            const int t = idx >> 6, e = idx & 63;
            const float z = s_xz[t * (2 * ED_) + ED_ + e];
            s_yg[idx] = (_Float16)(s_y[idx] * z * fast_sigmoid(z));
        }
        __syncthreads();

        // ---- out-projection (16x64)@(64x32) + residual, waves 0..1 ----
        if (wave < 2) {
            const int col  = wave * 16 + (lane & 15);
            const int mrow = lane & 15;
            const int kb   = (lane < 16) ? 0 : 8;
            v8f acc = {};
#pragma unroll
            for (int kc = 0; kc < 2; ++kc) {
                v16h a;
#pragma unroll
                for (int j = 0; j < 8; ++j) {
                    a[j]     = s_yg[mrow * ED_ + kc * 32 + kb + j];
                    a[8 + j] = s_yg[mrow * ED_ + kc * 32 + kb + 16 + j];
                }
                const v16h bfrag =
                    *(const v16h*)&s_outwf[((kc * 2 + wave) * 32 + lane) * 16];
                acc = __builtin_amdgcn_wmma_f32_16x16x32_f16(
                    false, a, false, bfrag, (short)0, acc, false, false);
            }
            const int rbase = (lane < 16) ? 0 : 8;
#pragma unroll
            for (int vr = 0; vr < 8; ++vr) {
                const int r = vr + rbase;
                xm[(t0 + r) * D_ + col] = acc[vr] + s_xch[r * D_ + col];
            }
        }
        __syncthreads();
    }
}

// ---------------------------------------------------------------------------
// Stage 3: head at t = T-1. FFN + residual, LayerNorm, fuse with the
// structurally-zero attention branch (x_attn_last == ln1_b), final fc.
// One wave per (b,l).
// ---------------------------------------------------------------------------
__global__ void __launch_bounds__(256) head_kernel(
    const float* __restrict__ XM,
    const float* __restrict__ w1, const float* __restrict__ b1,
    const float* __restrict__ w2, const float* __restrict__ b2,
    const float* __restrict__ ln2g, const float* __restrict__ ln2b,
    const float* __restrict__ ln1b, const float* __restrict__ fgate,
    const float* __restrict__ fcw,  const float* __restrict__ fcb,
    float* __restrict__ out)
{
    const int tid = threadIdx.x, lane = tid & 31, wave = tid >> 5;
    const int n = blockIdx.x * 8 + wave;
    if (n >= NSEQ) return;
    const int l = n % L_;
    const int d = lane;

    const float xv = XM[((long)n * T_ + (T_ - 1)) * D_ + d];

    float h1 = b1[d];
#pragma unroll
    for (int k = 0; k < D_; ++k) h1 += __shfl(xv, k, 32) * w1[k * D_ + d];
    h1 = fmaxf(h1, 0.f);

    float ffp = h1 * w2[d];
#pragma unroll
    for (int off = 16; off; off >>= 1) ffp += __shfl_xor(ffp, off, 32);
    const float v = xv + (ffp + b2[0]);

    float mu = v;
#pragma unroll
    for (int off = 16; off; off >>= 1) mu += __shfl_xor(mu, off, 32);
    mu *= (1.0f / D_);
    const float dv = v - mu;
    float var = dv * dv;
#pragma unroll
    for (int off = 16; off; off >>= 1) var += __shfl_xor(var, off, 32);
    var *= (1.0f / D_);
    const float vn = dv * rsqrtf(var + 1e-5f) * ln2g[d] + ln2b[d];

    const float g = fast_sigmoid(fgate[l]);
    const float fused = g * ln1b[d] + (1.f - g) * vn;

    float op = fused * fcw[d];
#pragma unroll
    for (int off = 16; off; off >>= 1) op += __shfl_xor(op, off, 32);
    if (lane == 0) out[n] = op + fcb[0];
}

// ---------------------------------------------------------------------------
extern "C" void kernel_launch(void* const* d_in, const int* in_sizes, int n_in,
                              void* d_out, int out_size, void* d_ws, size_t ws_size,
                              hipStream_t stream) {
    (void)in_sizes; (void)n_in; (void)out_size; (void)ws_size;
    const float* x        = (const float*)d_in[0];
    const float* emb_w    = (const float*)d_in[1];
    const float* emb_b    = (const float*)d_in[2];
    const float* pos_emb  = (const float*)d_in[3];
    const float* agg_w    = (const float*)d_in[4];
    const float* agg_b    = (const float*)d_in[5];
    // d_in[6] sim_w, d_in[7] ln1_g : structurally unused (attention branch dead)
    const float* ln1_b    = (const float*)d_in[8];
    const float* m_norm_w = (const float*)d_in[9];
    const float* m_in_w   = (const float*)d_in[10];
    const float* m_conv_w = (const float*)d_in[11];
    const float* m_conv_b = (const float*)d_in[12];
    const float* m_x_w    = (const float*)d_in[13];
    const float* m_dt_w   = (const float*)d_in[14];
    const float* m_dt_b   = (const float*)d_in[15];
    const float* m_A_log  = (const float*)d_in[16];
    const float* m_D      = (const float*)d_in[17];
    const float* m_out_w  = (const float*)d_in[18];
    const float* ffn_w1   = (const float*)d_in[19];
    const float* ffn_b1   = (const float*)d_in[20];
    const float* ffn_w2   = (const float*)d_in[21];
    const float* ffn_b2   = (const float*)d_in[22];
    const float* ln2_g    = (const float*)d_in[23];
    const float* ln2_b    = (const float*)d_in[24];
    const float* fgate    = (const float*)d_in[25];
    const float* fc_w     = (const float*)d_in[26];
    const float* fc_b     = (const float*)d_in[27];

    float* XM = (float*)d_ws;   // (B*L, T, D) residual stream, 27 MB

    const int npos_blocks = (B_ * T_ * L_) / 8;   // 26352, exact
    embed_agg_kernel<<<npos_blocks, 256, 0, stream>>>(
        x, emb_w, emb_b, pos_emb, agg_w, agg_b, XM);

    for (int i = 0; i < 2; ++i) {
        mamba_layer_kernel<<<NSEQ, 256, 0, stream>>>(
            XM,
            m_norm_w + i * D_,
            m_in_w   + (size_t)i * D_ * 2 * ED_,
            m_conv_w + i * ED_ * DC_,
            m_conv_b + i * ED_,
            m_x_w    + i * ED_ * DBCW,
            m_dt_w   + i * 2 * ED_,
            m_dt_b   + i * ED_,
            m_A_log  + i * ED_ * DS_,
            m_D      + i * ED_,
            m_out_w  + i * ED_ * D_);
    }

    head_kernel<<<(NSEQ + 7) / 8, 256, 0, stream>>>(
        XM, ffn_w1, ffn_b1, ffn_w2, ffn_b2, ln2_g, ln2_b,
        ln1_b, fgate, fc_w, fc_b, (float*)d_out);
}
